// Atom91Head_30580167147700
// MI455X (gfx1250) — matrix operand order repaired and checked by reading, not compiled, not run on hardware
//
#include <hip/hip_runtime.h>

// ---------------------------------------------------------------------------
// Atom91Head for MI455X (gfx1250, wave32, WMMA).
//  * edge_rot cancels exactly (RᵀR = I, per-channel scales, shared l=1 weight).
//  * Only coll[:, :4, :] is consumed downstream -> conv computes m = 0..3.
//  * Dense contractions: v_wmma_f32_16x16x32_bf16, A via float4 global loads,
//    B pre-staged to LDS as zero-padded bf16 (ds_load_b128 in the hot loop).
//  * (N,4,91) activations stored with stride 92/368 for 16B alignment.
// ---------------------------------------------------------------------------

typedef __attribute__((ext_vector_type(16))) __bf16 v16bf;
typedef __attribute__((ext_vector_type(8)))  float  v8f;

#define Nn 10000
#define Ee 60000
#define NS 368   // per-node stride of (4 x 91->92) activation rows
#define MS 92    // per-m stride

__device__ inline v8f wmma_bf16(v16bf a, v16bf b, v8f c) {
  return __builtin_amdgcn_wmma_f32_16x16x32_bf16(false, a, false, b, (short)0, c, false, false);
}

__device__ inline float4 ld4(const float* p) { return *(const float4*)p; }

__device__ inline void maskf4(float4& a, int kb, int K) {
  a.x = (kb + 0 < K) ? a.x : 0.f;
  a.y = (kb + 1 < K) ? a.y : 0.f;
  a.z = (kb + 2 < K) ? a.z : 0.f;
  a.w = (kb + 3 < K) ? a.w : 0.f;
}

__device__ inline v16bf cvt16(float4 a0, float4 a1, float4 a2, float4 a3) {
  v16bf v;
  v[0]  = (__bf16)a0.x; v[1]  = (__bf16)a0.y; v[2]  = (__bf16)a0.z; v[3]  = (__bf16)a0.w;
  v[4]  = (__bf16)a1.x; v[5]  = (__bf16)a1.y; v[6]  = (__bf16)a1.z; v[7]  = (__bf16)a1.w;
  v[8]  = (__bf16)a2.x; v[9]  = (__bf16)a2.y; v[10] = (__bf16)a2.z; v[11] = (__bf16)a2.w;
  v[12] = (__bf16)a3.x; v[13] = (__bf16)a3.y; v[14] = (__bf16)a3.z; v[15] = (__bf16)a3.w;
  return v;
}

// A fragment (16x32 bf16, ISA 7.12.2 layout) from an aligned row-major f32 row.
// rowbase must be 16B aligned; row must have ceil32(K) floats accessible.
__device__ inline v16bf load_a_rm(const float* rowbase, int k0, int K) {
  int lane = threadIdx.x & 31;
  int hk = (lane & 16) ? 8 : 0;
  const float* p = rowbase + k0 + hk;
  float4 a0 = ld4(p);      float4 a1 = ld4(p + 4);
  float4 a2 = ld4(p + 16); float4 a3 = ld4(p + 20);
  if (k0 + 32 > K) {                 // wave-uniform tail branch, cndmask inside
    int b0 = k0 + hk;
    maskf4(a0, b0, K); maskf4(a1, b0 + 4, K);
    maskf4(a2, b0 + 16, K); maskf4(a3, b0 + 20, K);
  }
  return cvt16(a0, a1, a2, a3);
}

// Stage W (K x ncol, row-major f32) into LDS as bf16, layout lw[col*Kpad + k],
// zero-padded to Kpad x colpad. Caller must __syncthreads() afterwards.
__device__ inline void stage_w(const float* __restrict__ W, int K, int ldw, int ncol,
                               int Kpad, int colpad, __bf16* lw) {
  for (int idx = threadIdx.x; idx < Kpad * colpad; idx += blockDim.x) {
    int k = idx / colpad, c = idx % colpad;
    float v = (k < K && c < ncol) ? W[(long)k * ldw + c] : 0.0f;
    lw[(long)c * Kpad + k] = (__bf16)v;
  }
}

// B fragment (32x16 bf16): lane col = ct*16+(lane&15); elems 0-7 = k0+hk.., 8-15 = k0+16+hk.. (hk = 8*(lane/16)).
__device__ inline v16bf load_b_lds(const __bf16* lw, int Kpad, int ct, int k0) {
  typedef __attribute__((ext_vector_type(8))) __bf16 v8bf_t;
  int lane = threadIdx.x & 31;
  const __bf16* p = lw + (long)(ct * 16 + (lane & 15)) * Kpad + k0 + ((lane & 16) ? 8 : 0);
  v8bf_t lo = *(const v8bf_t*)p;          // 16B aligned -> ds_load_b128
  v8bf_t hi = *(const v8bf_t*)(p + 16);   // 16B aligned -> ds_load_b128
  return __builtin_shufflevector(lo, hi, 0, 1, 2, 3, 4, 5, 6, 7, 8, 9, 10, 11, 12, 13, 14, 15);
}

__device__ inline unsigned ord_f32(float f) {
  unsigned u = __float_as_uint(f);
  return (u & 0x80000000u) ? ~u : (u | 0x80000000u);
}
__device__ inline float unord_f32(unsigned u) {
  return (u & 0x80000000u) ? __uint_as_float(u ^ 0x80000000u) : __uint_as_float(~u);
}
__device__ inline float silu_f(float x) { return x / (1.0f + __expf(-x)); }

__device__ __constant__ unsigned char d_reset[91] = {
  1,1,1,1, 1,0,0, 1,0,0,0, 1,0,0,0,0, 1,0,0,0, 1,0,0,0,0,0, 1,0,0,0,
  1,0,0,0,0, 1,0,0,0, 1,0,0, 1,0,0,0,0, 1,0,0,0, 1,0,0,0,0, 1,0,0,0,
  1,0,0, 1,0,0,0,0, 1,0,0,0, 1,0,0,0,0,0, 1,0,0,0, 1,0,0,0,0, 1,0,0,0
};

__global__ void k_zero(float* p, long n) {
  long i = (long)blockIdx.x * blockDim.x + threadIdx.x;
  long s = (long)gridDim.x * blockDim.x;
  for (; i < n; i += s) p[i] = 0.0f;
}

// ------------------- generic WMMA GEMM (nrows % 16 == 0) -------------------
// flags: bit0 = silu, bit1 = accumulate
__global__ void k_gemm(const float* __restrict__ X, int ldx, int nrows,
                       const float* __restrict__ W, int K, int ncol, int ldw,
                       const float* __restrict__ bias,
                       float* __restrict__ Y, int ldy, int flags) {
  extern __shared__ __bf16 lw[];
  int Kpad = (K + 31) & ~31;
  int colpad = (ncol + 15) & ~15;
  stage_w(W, K, ldw, ncol, Kpad, colpad, lw);
  __syncthreads();
  int wid  = blockIdx.x * (blockDim.x >> 5) + (threadIdx.x >> 5);
  int lane = threadIdx.x & 31;
  int colT = colpad >> 4;
  if (wid >= (nrows >> 4) * colT) return;
  int rt = wid / colT, ct = wid % colT;
  const float* rowbase = X + (long)(rt * 16 + (lane & 15)) * ldx;
  v8f acc = {0.f, 0.f, 0.f, 0.f, 0.f, 0.f, 0.f, 0.f};
  for (int k0 = 0; k0 < K; k0 += 32)
    acc = wmma_bf16(load_a_rm(rowbase, k0, K), load_b_lds(lw, Kpad, ct, k0), acc);
  int col = ct * 16 + (lane & 15);
  if (col < ncol) {
    float bb = bias ? bias[col] : 0.0f;
#pragma unroll
    for (int r = 0; r < 8; ++r) {
      int row = rt * 16 + r + ((lane & 16) ? 8 : 0);
      float v = acc[r] + bb;
      if (flags & 1) v = silu_f(v);
      if (flags & 2) Y[(long)row * ldy + col] += v;
      else           Y[(long)row * ldy + col]  = v;
    }
  }
}

// ------------------- fuse: [density[:,0,:], seq] @ W_fuse + b --------------
__global__ void k_fuse(const float* __restrict__ dens, const float* __restrict__ seq,
                       const float* __restrict__ Wf, const float* __restrict__ bf,
                       float* __restrict__ fused) {
  __shared__ __bf16 lw[448 * 64];
  stage_w(Wf, 448, 64, 64, 448, 64, lw);
  __syncthreads();
  int wid  = blockIdx.x * (blockDim.x >> 5) + (threadIdx.x >> 5);
  int lane = threadIdx.x & 31;
  if (wid >= 625 * 4) return;
  int rt = wid >> 2, ct = wid & 3;
  int n = rt * 16 + (lane & 15);
  const float* db = dens + (long)n * 1600;
  const float* sb = seq + (long)n * 384 - 64;   // so rowbase[k] = seq[k-64]
  v8f acc = {0.f, 0.f, 0.f, 0.f, 0.f, 0.f, 0.f, 0.f};
#pragma unroll
  for (int k0 = 0; k0 < 448; k0 += 32) {
    const float* rb = (k0 >= 64) ? sb : db;     // chunk regions never straddle 64
    acc = wmma_bf16(load_a_rm(rb, k0, 448), load_b_lds(lw, 448, ct, k0), acc);
  }
  int col = ct * 16 + (lane & 15);
#pragma unroll
  for (int r = 0; r < 8; ++r) {
    int row = rt * 16 + r + ((lane & 16) ? 8 : 0);
    fused[(long)row * 64 + col] = acc[r] + bf[col];
  }
}

// ------------------- conv: coll[dst,m] += (x[src,m] .* gate) @ Wconv[l(m)] --
__global__ void k_conv(const float* __restrict__ fused, const float* __restrict__ dens,
                       const float* __restrict__ gate, const float* __restrict__ Wc,
                       const int* __restrict__ srcI, const int* __restrict__ dstI,
                       float* __restrict__ coll) {
  __shared__ __bf16 lw[2 * 96 * 64];
  stage_w(Wc, 64, 91, 91, 64, 96, lw);
  stage_w(Wc + 64 * 91, 64, 91, 91, 64, 96, lw + 96 * 64);
  __syncthreads();
  int wid  = blockIdx.x * (blockDim.x >> 5) + (threadIdx.x >> 5);
  int lane = threadIdx.x & 31;
  if (wid >= 3750 * 24) return;
  int rt = wid / 24, rem = wid % 24;
  int m = rem / 6, ct = rem % 6;
  const __bf16* lwm = lw + (m ? 1 : 0) * 96 * 64;
  int e = rt * 16 + (lane & 15);
  int s = srcI[e];
  const float* xb = (m == 0) ? (fused + (long)s * 64) : (dens + (long)s * 1600 + m * 64);
  const float* gb = gate + (long)e * 64;
  int hk = (lane & 16) ? 8 : 0;
  v8f acc = {0.f, 0.f, 0.f, 0.f, 0.f, 0.f, 0.f, 0.f};
#pragma unroll
  for (int k0 = 0; k0 < 64; k0 += 32) {
    const float* xp = xb + k0 + hk;
    const float* gp = gb + k0 + hk;
    float4 x0 = ld4(xp),      x1 = ld4(xp + 4),  x2 = ld4(xp + 16), x3 = ld4(xp + 20);
    float4 g0 = ld4(gp),      g1 = ld4(gp + 4),  g2 = ld4(gp + 16), g3 = ld4(gp + 20);
    x0.x *= g0.x; x0.y *= g0.y; x0.z *= g0.z; x0.w *= g0.w;
    x1.x *= g1.x; x1.y *= g1.y; x1.z *= g1.z; x1.w *= g1.w;
    x2.x *= g2.x; x2.y *= g2.y; x2.z *= g2.z; x2.w *= g2.w;
    x3.x *= g3.x; x3.y *= g3.y; x3.z *= g3.z; x3.w *= g3.w;
    acc = wmma_bf16(cvt16(x0, x1, x2, x3), load_b_lds(lwm, 64, ct, k0), acc);
  }
  int col = ct * 16 + (lane & 15);
  if (col < 91) {
#pragma unroll
    for (int r = 0; r < 8; ++r) {
      int ee = rt * 16 + r + ((lane & 16) ? 8 : 0);
      atomicAdd(&coll[(long)dstI[ee] * NS + m * MS + col], acc[r]);
    }
  }
}

// ------------------- post-conv: +b_conv, so3_norm(l0,l1), cumsum -----------
__global__ void k_postconv(const float* __restrict__ coll, const float* __restrict__ bconv,
                           const float* __restrict__ g, float* __restrict__ a) {
  int n = blockIdx.x * blockDim.x + threadIdx.x;
  if (n >= Nn) return;
  const float* cp = coll + (long)n * NS;
  float ss0 = 0.f, ss1 = 0.f;
  for (int c = 0; c < 91; ++c) { float v = cp[c] + bconv[c]; ss0 += v * v; }
  for (int m = 1; m < 4; ++m)
    for (int c = 0; c < 91; ++c) { float v = cp[m * MS + c]; ss1 += v * v; }
  float r0 = rsqrtf(ss0 / 91.0f + 1e-6f);
  float r1 = rsqrtf(ss1 / 273.0f + 1e-6f);
  float* ap = a + (long)n * NS;
  for (int m = 0; m < 4; ++m) {
    float run = 0.f;
    for (int c = 0; c < 91; ++c) {
      float v = (m == 0) ? (cp[c] + bconv[c]) * r0 * g[c]
                         : cp[m * MS + c] * r1 * g[91 + c];
      run = d_reset[c] ? v : (run + v);
      ap[m * MS + c] = run;
    }
  }
}

// ------------------- so3_norm over OFF4 blocks -----------------------------
__global__ void k_so3norm4(const float* __restrict__ X, const float* __restrict__ g,
                           float* __restrict__ Y) {
  int n = blockIdx.x * blockDim.x + threadIdx.x;
  if (n >= Nn) return;
  const float* xp = X + (long)n * NS;
  float ss0 = 0.f, ss1 = 0.f;
  for (int c = 0; c < 91; ++c) { float v = xp[c]; ss0 += v * v; }
  for (int m = 1; m < 4; ++m)
    for (int c = 0; c < 91; ++c) { float v = xp[m * MS + c]; ss1 += v * v; }
  float r0 = rsqrtf(ss0 / 91.0f + 1e-6f);
  float r1 = rsqrtf(ss1 / 273.0f + 1e-6f);
  float* yp = Y + (long)n * NS;
  for (int c = 0; c < 91; ++c) yp[c] = xp[c] * r0 * g[c];
  for (int m = 1; m < 4; ++m)
    for (int c = 0; c < 91; ++c) yp[m * MS + c] = xp[m * MS + c] * r1 * g[91 + c];
}

// ------------------- attention layer-1: silu([hs0,hd0,ef] @ Wa1 + ba1) -----
__global__ void k_att1(const float* __restrict__ h, const float* __restrict__ ef,
                       const float* __restrict__ Wa1, const float* __restrict__ ba1,
                       const int* __restrict__ srcI, const int* __restrict__ dstI,
                       float* __restrict__ t) {
  __shared__ __bf16 lw[256 * 32];
  stage_w(Wa1, 246, 32, 32, 256, 32, lw);
  __syncthreads();
  int wid  = blockIdx.x * (blockDim.x >> 5) + (threadIdx.x >> 5);
  int lane = threadIdx.x & 31;
  if (wid >= 3750 * 2) return;
  int rt = wid >> 1, ct = wid & 1;
  int e = rt * 16 + (lane & 15);
  int s = srcI[e], d = dstI[e];
  const float* hs = h + (long)s * NS;
  const float* hd = h + (long)d * NS;
  const float* eb = ef + (long)e * 64;
  v8f acc = {0.f, 0.f, 0.f, 0.f, 0.f, 0.f, 0.f, 0.f};
  for (int k0 = 0; k0 < 246; k0 += 32) {
    v16bf av;
#pragma unroll
    for (int i = 0; i < 16; ++i) {
      int k = k0 + ((i < 8) ? 0 : 16) + ((lane & 16) ? 8 : 0) + (i & 7);
      // branchless pointer select; all addresses in-bounds
      const float* ph = (k < 91) ? (hs + k) : (hd + (k - 91));
      const float* pe = eb + ((k < 246) ? (k - 182) : 63);
      const float* p  = (k < 182) ? ph : pe;
      float v = *p;
      av[i] = (__bf16)((k < 246) ? v : 0.f);
    }
    acc = wmma_bf16(av, load_b_lds(lw, 256, ct, k0), acc);
  }
  int col = ct * 16 + (lane & 15);
  float bb = ba1[col];
#pragma unroll
  for (int r = 0; r < 8; ++r) {
    int ee = rt * 16 + r + ((lane & 16) ? 8 : 0);
    t[(long)ee * 32 + col] = silu_f(acc[r] + bb);
  }
}

// ------------------- segment softmax pieces --------------------------------
__global__ void k_segmax(const float* __restrict__ logits, const int* __restrict__ dstI,
                         unsigned* __restrict__ mbuf) {
  int t = blockIdx.x * blockDim.x + threadIdx.x;
  if (t >= Ee * 8) return;
  int e = t >> 3, hh = t & 7;
  atomicMax(&mbuf[(long)dstI[e] * 8 + hh], ord_f32(logits[t]));
}
__global__ void k_exps(float* __restrict__ exb, const int* __restrict__ dstI,
                       const unsigned* __restrict__ mbuf, float* __restrict__ sbuf) {
  int t = blockIdx.x * blockDim.x + threadIdx.x;
  if (t >= Ee * 8) return;
  int e = t >> 3, hh = t & 7;
  int d = dstI[e];
  float ev = __expf(exb[t] - unord_f32(mbuf[(long)d * 8 + hh]));
  exb[t] = ev;
  atomicAdd(&sbuf[(long)d * 8 + hh], ev);
}

// ------------------- value: agg[dst,m] += alpha .* (h[src,m] @ Wv[l(m)]) ----
__global__ void k_value(const float* __restrict__ h, const float* __restrict__ Wv2,
                        const int* __restrict__ srcI, const int* __restrict__ dstI,
                        const float* __restrict__ exb, const float* __restrict__ sbuf,
                        float* __restrict__ agg) {
  __shared__ __bf16 lw[2 * 96 * 64];
  stage_w(Wv2, 91, 64, 64, 96, 64, lw);
  stage_w(Wv2 + 91 * 64, 91, 64, 64, 96, 64, lw + 96 * 64);
  __syncthreads();
  int wid  = blockIdx.x * (blockDim.x >> 5) + (threadIdx.x >> 5);
  int lane = threadIdx.x & 31;
  if (wid >= 3750 * 16) return;
  int rt = wid >> 4, rem = wid & 15;
  int m = rem >> 2, ct = rem & 3;
  const __bf16* lwm = lw + (m ? 1 : 0) * 96 * 64;
  int e = rt * 16 + (lane & 15);
  const float* rowbase = h + (long)srcI[e] * NS + m * MS;
  v8f acc = {0.f, 0.f, 0.f, 0.f, 0.f, 0.f, 0.f, 0.f};
  for (int k0 = 0; k0 < 91; k0 += 32)
    acc = wmma_bf16(load_a_rm(rowbase, k0, 91), load_b_lds(lwm, 96, ct, k0), acc);
  int col  = ct * 16 + (lane & 15);
  int head = col >> 3;
#pragma unroll
  for (int r = 0; r < 8; ++r) {
    int ee = rt * 16 + r + ((lane & 16) ? 8 : 0);
    int d  = dstI[ee];
    float alpha = exb[(long)ee * 8 + head] / (sbuf[(long)d * 8 + head] + 1e-9f);
    atomicAdd(&agg[(long)d * 256 + m * 64 + col], acc[r] * alpha);
  }
}

// ------------------- FFN gating --------------------------------------------
__global__ void k_ffngate(float* __restrict__ hid, const float* __restrict__ bf1) {
  int t = blockIdx.x * blockDim.x + threadIdx.x;
  if (t >= Nn * 32) return;
  int n = t >> 5, c = t & 31;
  float* hp = hid + (long)n * 128;
  float s0 = silu_f(hp[c] + bf1[c]);
  hp[c] = s0;
  float sg = 1.0f / (1.0f + __expf(-s0));
  hp[32 + c] *= sg; hp[64 + c] *= sg; hp[96 + c] *= sg;
}

// ---------------------------------------------------------------------------
extern "C" void kernel_launch(void* const* d_in, const int* in_sizes, int n_in,
                              void* d_out, int out_size, void* d_ws, size_t ws_size,
                              hipStream_t stream) {
  const float* dens  = (const float*)d_in[0];
  const float* seq   = (const float*)d_in[1];
  const float* ef    = (const float*)d_in[2];
  /* d_in[3] edge_rot: unused — rotations cancel (RᵀR = I) */
  const float* Wfuse = (const float*)d_in[4];
  const float* bfuse = (const float*)d_in[5];
  const float* Wr1   = (const float*)d_in[6];
  const float* br1   = (const float*)d_in[7];
  const float* Wr2   = (const float*)d_in[8];
  const float* br2   = (const float*)d_in[9];
  const float* Wconv = (const float*)d_in[10];
  const float* bconv = (const float*)d_in[11];
  const float* gcn   = (const float*)d_in[12];
  const float* gln1  = (const float*)d_in[13];
  const float* Wa1   = (const float*)d_in[14];
  const float* ba1   = (const float*)d_in[15];
  const float* Wa2   = (const float*)d_in[16];
  const float* ba2   = (const float*)d_in[17];
  const float* Wv    = (const float*)d_in[18];
  const float* Wo    = (const float*)d_in[19];
  const float* gln2  = (const float*)d_in[20];
  const float* Wf1   = (const float*)d_in[21];
  const float* bf1   = (const float*)d_in[22];
  const float* Wf2   = (const float*)d_in[23];
  const float* bf2   = (const float*)d_in[24];
  const float* Wp    = (const float*)d_in[25];
  const float* bp    = (const float*)d_in[26];
  const int*   ei    = (const int*)d_in[27];
  const int* srcI = ei;
  const int* dstI = ei + Ee;
  float* out = (float*)d_out;

  float* ws    = (float*)d_ws;
  float* fused = ws;                   // N*64
  float* gate  = fused + 640000;       // E*64
  float* hb    = gate + 3840000;       // E*64 (gate hidden, then attn hidden)
  float* coll  = hb + 3840000;         // N*368 (+pad)
  float* abuf  = coll + 3680064;       // N*368 (+pad)
  float* hbuf  = abuf + 3680064;       // N*368 (+pad)
  float* exb   = hbuf + 3680064;       // E*8
  float* msb   = exb + 480000;         // N*8 mbuf + N*8 sbuf
  unsigned* mbuf = (unsigned*)msb;
  float* sbuf  = msb + 80000;
  float* agg   = msb + 160000;         // N*4*64 (+pad)
  float* hid   = agg + 2560064;        // N*4*32

  auto blk = [](int waves) { return (waves + 7) / 8; };
  auto ldsB = [](int K, int ncol) {
    return (size_t)(((K + 31) & ~31) * ((ncol + 15) & ~15) * 2);
  };

  // Stage A: fuse
  k_fuse<<<blk(625 * 4), 256, 0, stream>>>(dens, seq, Wfuse, bfuse, fused);
  // Stage B: gate = silu(ef@Wr1+br1)@Wr2+br2
  k_gemm<<<blk(3750 * 4), 256, ldsB(64, 64), stream>>>(ef, 64, Ee, Wr1, 64, 64, 64, br1, hb, 64, 1);
  k_gemm<<<blk(3750 * 4), 256, ldsB(64, 64), stream>>>(hb, 64, Ee, Wr2, 64, 64, 64, br2, gate, 64, 0);
  // Stage C: conv (rows 0..3 only)
  k_zero<<<256, 256, 0, stream>>>(coll, 3680064);
  k_conv<<<blk(3750 * 24), 256, 0, stream>>>(fused, dens, gate, Wconv, srcI, dstI, coll);
  // Stage D: +b_conv, so3_norm, cumsum
  k_postconv<<<(Nn + 255) / 256, 256, 0, stream>>>(coll, bconv, gcn, abuf);

  for (int i = 0; i < 3; ++i) {
    k_so3norm4<<<(Nn + 255) / 256, 256, 0, stream>>>(abuf, gln1 + i * 182, hbuf);
    k_att1<<<blk(3750 * 2), 256, 0, stream>>>(hbuf, ef, Wa1 + i * 246 * 32, ba1 + i * 32,
                                              srcI, dstI, hb);
    k_gemm<<<blk(3750), 256, ldsB(32, 8), stream>>>(hb, 32, Ee, Wa2 + i * 256, 32, 8, 8,
                                                    ba2 + i * 8, exb, 8, 0);
    k_zero<<<64, 256, 0, stream>>>(msb, 160000);
    k_segmax<<<(Ee * 8 + 255) / 256, 256, 0, stream>>>(exb, dstI, mbuf);
    k_exps<<<(Ee * 8 + 255) / 256, 256, 0, stream>>>(exb, dstI, mbuf, sbuf);
    k_zero<<<256, 256, 0, stream>>>(agg, 2560064);
    k_value<<<blk(3750 * 16), 256, 0, stream>>>(hbuf, Wv + i * 2 * 91 * 64, srcI, dstI,
                                                exb, sbuf, agg);
    for (int m = 0; m < 4; ++m) {
      int lm = m ? 1 : 0;
      k_gemm<<<blk(625 * 6), 256, ldsB(64, 91), stream>>>(
          agg + m * 64, 256, Nn, Wo + i * 2 * 64 * 91 + lm * 64 * 91,
          64, 91, 91, nullptr, abuf + m * MS, NS, 2);
    }
    k_so3norm4<<<(Nn + 255) / 256, 256, 0, stream>>>(abuf, gln2 + i * 182, hbuf);
    for (int m = 0; m < 4; ++m) {
      int lm = m ? 1 : 0;
      k_gemm<<<blk(625 * 2), 256, ldsB(91, 32), stream>>>(
          hbuf + m * MS, NS, Nn, Wf1 + i * 2 * 91 * 32 + lm * 91 * 32,
          91, 32, 32, nullptr, hid + m * 32, 128, 0);
    }
    k_ffngate<<<(Nn * 32 + 255) / 256, 256, 0, stream>>>(hid, bf1 + i * 32);
    for (int m = 0; m < 4; ++m) {
      int lm = m ? 1 : 0;
      k_gemm<<<blk(625 * 6), 256, ldsB(32, 91), stream>>>(
          hid + m * 32, 128, Nn, Wf2 + i * 2 * 32 * 91 + lm * 32 * 91,
          32, 91, 91, (m == 0) ? (bf2 + i * 91) : nullptr, abuf + m * MS, NS, 2);
    }
  }
  // Final projection: y = a @ Wp[l(m)] (+bp on row 0); out keeps stride 364
  for (int m = 0; m < 4; ++m) {
    int lm = m ? 1 : 0;
    k_gemm<<<blk(625 * 6), 256, ldsB(91, 91), stream>>>(
        abuf + m * MS, NS, Nn, Wp + lm * 91 * 91, 91, 91, 91,
        (m == 0) ? bp : nullptr, out + m * 91, 364, 0);
  }
  (void)in_sizes; (void)n_in; (void)out_size; (void)ws_size;
}
